// MLPModel7_80109730005568
// MI455X (gfx1250) — matrix operand order repaired and checked by reading, not compiled, not run on hardware
//
#include <hip/hip_runtime.h>
#include <hip/hip_bf16.h>

// MI455X / gfx1250, wave32, WMMA f16 16x16x32 with f32 accumulate,
// async global->LDS staging (ASYNCcnt) for the shared GEMM operand.

#define NN 2048
#define DD 128
#define HH 256
#define KFLAT (DD*HH)   // 32768
#define XSTR 129        // padded LDS row stride (floats) -> conflict-free

typedef __attribute__((ext_vector_type(16))) _Float16 v16h;
typedef __attribute__((ext_vector_type(8)))  _Float16 v8h;
typedef __attribute__((ext_vector_type(4)))  _Float16 v4h;
typedef __attribute__((ext_vector_type(8)))  float    v8f;

#if __has_builtin(__builtin_amdgcn_tanhf)
#define TANHF(x) __builtin_amdgcn_tanhf(x)   // v_tanh_f32 (gfx1250 trans op)
#else
#define TANHF(x) tanhf(x)
#endif

static __device__ __forceinline__ v16h mk16(v8h lo, v8h hi) {
    v16h r;
#pragma unroll
    for (int i = 0; i < 8; ++i) { r[i] = lo[i]; r[i + 8] = hi[i]; }
    return r;
}

#define WMMA16(a, b, c) \
    __builtin_amdgcn_wmma_f32_16x16x32_f16(false, (a), false, (b), (short)0, (c), false, false)

// Async global -> LDS copy of 16 bytes per lane (ASYNCcnt-tracked).
// Low 32 bits of a generic shared pointer == LDS offset (aperture is in [63:32]).
static __device__ __forceinline__ void async_copy_b128(void* lds_ptr, const void* gptr) {
    unsigned int  ldsoff = (unsigned int)(unsigned long long)lds_ptr;
    unsigned long long ga = (unsigned long long)gptr;
    asm volatile("global_load_async_to_lds_b128 %0, %1, off"
                 :: "v"(ldsoff), "v"(ga) : "memory");
}
static __device__ __forceinline__ void async_wait_le4() {
    asm volatile("s_wait_asynccnt 0x4" ::: "memory");
}
static __device__ __forceinline__ void async_wait_0() {
    asm volatile("s_wait_asynccnt 0x0" ::: "memory");
}

// ---------------------------------------------------------------------------
// k_convert: repack Wh [32768,256] and Wf [128,128] (f32 row-major) into f16
// fragment-major layout: block b = ktile*NT + ntile; within block, lane's 16
// contiguous halves are its WMMA-B fragment (col = nt*16 + lane%16,
// k = ktile*32 + (j<8?0:16) + (lane<16?0:8) + (j&7)).
// ---------------------------------------------------------------------------
__global__ void k_convert(const float* __restrict__ Wh, const float* __restrict__ Wf,
                          _Float16* __restrict__ whf, _Float16* __restrict__ wff) {
    unsigned idx = blockIdx.x * 256u + threadIdx.x;
    const unsigned T1 = 32768u * 256u;  // whf halves
    if (idx < T1) {
        unsigned hh = idx & 15u, lane = (idx >> 4) & 31u, b = idx >> 9;
        unsigned nt = b & 15u, kt = b >> 4;
        unsigned col = nt * 16u + (lane & 15u);
        unsigned k   = kt * 32u + ((hh & 8u) << 1) + ((lane & 16u) >> 1) + (hh & 7u);
        whf[idx] = (_Float16)Wh[k * 256u + col];
    } else {
        unsigned i2 = idx - T1;
        if (i2 < 128u * 128u) {
            unsigned hh = i2 & 15u, lane = (i2 >> 4) & 31u, b = i2 >> 9;
            unsigned nt = b & 7u, kt = b >> 3;
            unsigned col = nt * 16u + (lane & 15u);
            unsigned k   = kt * 32u + ((hh & 8u) << 1) + ((lane & 16u) >> 1) + (hh & 7u);
            wff[i2] = (_Float16)Wf[k * 128u + col];
        }
    }
}

// ---------------------------------------------------------------------------
// k_attn: one workgroup (8 waves) per sample.
//   x = relu(a*W_in+b_in) in LDS (f32, padded) ->
//   WMMA GEMM u=tanh(x@Wf+bf), scores=u@Vf -> softmax over H ->
//   y = x*(1+attn), LayerNorm over D -> write f16 row-major (head-GEMM A layout)
// ---------------------------------------------------------------------------
__global__ void __launch_bounds__(256)
k_attn(const float* __restrict__ arr, const float* __restrict__ Win,
       const float* __restrict__ bin, const _Float16* __restrict__ wff,
       const float* __restrict__ bf, const float* __restrict__ Vf,
       const float* __restrict__ lng, const float* __restrict__ lnb,
       _Float16* __restrict__ xx) {
    __shared__ float a_s[DD];
    __shared__ float x_s[HH * XSTR];   // 132 KB
    __shared__ float scores[HH];
    __shared__ float red[HH];
    __shared__ float attn[HH];

    const int t = threadIdx.x;
    const int n = blockIdx.x;
    if (t < DD) a_s[t] = arr[n * DD + t];
    __syncthreads();

    // x[h=t][d] = relu(a[d]*W_in[d][t] + b_in[d][t]); coalesced over h
    {
        float* xr = &x_s[t * XSTR];
#pragma unroll 4
        for (int d = 0; d < DD; ++d) {
            float v = fmaf(a_s[d], Win[d * HH + t], bin[d * HH + t]);
            xr[d] = v > 0.f ? v : 0.f;
        }
    }
    __syncthreads();

    const int w = t >> 5, lane = t & 31, l16 = lane & 15;
    const int hiN = (lane & 16) ? 1 : 0;

    // GEMM: each wave owns 2 M-tiles x 8 N-tiles, K = 128 (4 steps of 32)
    for (int mt = w * 2; mt < w * 2 + 2; ++mt) {
        const float* xr = &x_s[(mt * 16 + l16) * XSTR];
        v16h A[4];
#pragma unroll
        for (int ks = 0; ks < 4; ++ks) {
            const int kb = ks * 32 + (hiN ? 8 : 0);
            v16h a;
#pragma unroll
            for (int e = 0; e < 16; ++e) {
                const int k = kb + e + (e >= 8 ? 8 : 0);   // ISA 16-bit A layout
                a[e] = (_Float16)xr[k];
            }
            A[ks] = a;
        }
        float sacc[8];
#pragma unroll
        for (int r = 0; r < 8; ++r) sacc[r] = 0.f;

        for (int nt = 0; nt < 8; ++nt) {
            v8f c;
#pragma unroll
            for (int r = 0; r < 8; ++r) c[r] = 0.f;
#pragma unroll
            for (int ks = 0; ks < 4; ++ks) {
                const v8h* bp = (const v8h*)(wff + (((ks * 8 + nt) * 32 + lane) << 4));
                c = WMMA16(A[ks], mk16(bp[0], bp[1]), c);
            }
            const int col = nt * 16 + l16;
            const float bfv = bf[col], vfv = Vf[col];
#pragma unroll
            for (int r = 0; r < 8; ++r) sacc[r] += TANHF(c[r] + bfv) * vfv;
        }
        // reduce across the 16-lane half (sum over N), lanes 0/16 hold rows
#pragma unroll
        for (int r = 0; r < 8; ++r) {
            float v = sacc[r];
            v += __shfl_xor(v, 1, 32);  v += __shfl_xor(v, 2, 32);
            v += __shfl_xor(v, 4, 32);  v += __shfl_xor(v, 8, 32);
            if (l16 == 0) scores[mt * 16 + (hiN ? 8 : 0) + r] = v;
        }
    }
    __syncthreads();

    // softmax over H=256 (one score per thread)
    float s = scores[t];
    red[t] = s; __syncthreads();
    for (int st = 128; st > 0; st >>= 1) {
        if (t < st) red[t] = fmaxf(red[t], red[t + st]);
        __syncthreads();
    }
    const float mx = red[0];
    __syncthreads();
    const float e = __expf(s - mx);
    red[t] = e; __syncthreads();
    for (int st = 128; st > 0; st >>= 1) {
        if (t < st) red[t] += red[t + st];
        __syncthreads();
    }
    attn[t] = e / red[0];
    __syncthreads();

    // y = x*(1+attn); LayerNorm over D; store f16 row-major
    for (int i = 0; i < 32; ++i) {
        const int h = w * 32 + i;
        const float ah = 1.f + attn[h];
        const float* xr = &x_s[h * XSTR + lane * 4];
        float y0 = xr[0] * ah, y1 = xr[1] * ah, y2 = xr[2] * ah, y3 = xr[3] * ah;
        float sm = y0 + y1 + y2 + y3;
        float sq = y0 * y0 + y1 * y1 + y2 * y2 + y3 * y3;
#pragma unroll
        for (int m2 = 1; m2 < 32; m2 <<= 1) {
            sm += __shfl_xor(sm, m2, 32);
            sq += __shfl_xor(sq, m2, 32);
        }
        const float mean = sm * (1.f / 128.f);
        const float var  = sq * (1.f / 128.f) - mean * mean;
        const float rstd = rsqrtf(var + 1e-5f);
        const int d0 = lane * 4;
        v4h o;
        o[0] = (_Float16)((y0 - mean) * rstd * lng[d0 + 0] + lnb[d0 + 0]);
        o[1] = (_Float16)((y1 - mean) * rstd * lng[d0 + 1] + lnb[d0 + 1]);
        o[2] = (_Float16)((y2 - mean) * rstd * lng[d0 + 2] + lnb[d0 + 2]);
        o[3] = (_Float16)((y3 - mean) * rstd * lng[d0 + 3] + lnb[d0 + 3]);
        *(v4h*)(xx + (size_t)n * KFLAT + h * DD + d0) = o;
    }
}

// ---------------------------------------------------------------------------
// k_head: [2048 x 32768] @ [32768 x 256], grid (16 M-tiles x 16 K-chunks).
// B chunks (16 KB / K-step, shared by all 8 waves) are staged into LDS with
// double-buffered global_load_async_to_lds_b128 (ASYNCcnt), WMMAs read them
// via ds_load_b128. A fragments are per-wave private -> direct global loads.
// Partial sums merged via global f32 atomics.
// ---------------------------------------------------------------------------
__global__ void __launch_bounds__(256)
k_head(const _Float16* __restrict__ xx, const _Float16* __restrict__ whf,
       float* __restrict__ hacc) {
    __shared__ _Float16 bsh[2][8192];   // 2 x 16 KB double buffer

    const int t = threadIdx.x, w = t >> 5, lane = t & 31, l16 = lane & 15;
    const int hiN = (lane & 16) ? 1 : 0;
    const int mb = blockIdx.x, kc = blockIdx.y;
    const int row = mb * 128 + w * 16 + l16;
    const _Float16* xrow = xx + (size_t)row * KFLAT;
    const _Float16* chunk0 = whf + (size_t)(kc * 64) * 8192;  // 64 K-steps/chunk

    v8f acc[16];
#pragma unroll
    for (int i = 0; i < 16; ++i)
#pragma unroll
        for (int r = 0; r < 8; ++r) acc[i][r] = 0.f;

    // stage step 0 into buffer 0: 16 KB = 256 threads x 4 x 16 B
#pragma unroll
    for (int i = 0; i < 4; ++i)
        async_copy_b128(&bsh[0][(t + i * 256) * 8], chunk0 + (t + i * 256) * 8);

    for (int ks = 0; ks < 64; ++ks) {
        const int cur = ks & 1;
        if (ks < 63) {
#pragma unroll
            for (int i = 0; i < 4; ++i)
                async_copy_b128(&bsh[cur ^ 1][(t + i * 256) * 8],
                                chunk0 + (size_t)(ks + 1) * 8192 + (t + i * 256) * 8);
            async_wait_le4();   // 8 in flight; in-order completion => cur's 4 landed
        } else {
            async_wait_0();
        }
        __syncthreads();

        const int kk  = kc * 2048 + ks * 32;
        const int kbA = kk + (hiN ? 8 : 0);
        const v8h alo = *(const v8h*)(xrow + kbA);
        const v8h ahi = *(const v8h*)(xrow + kbA + 16);
        const v16h af = mk16(alo, ahi);
        if (ks < 63) __builtin_prefetch(xrow + kbA + 32, 0, 1);

        const _Float16* bbase = &bsh[cur][lane * 16];
#pragma unroll
        for (int nt = 0; nt < 16; ++nt) {
            const v8h* bp = (const v8h*)(bbase + nt * 512);
            acc[nt] = WMMA16(af, mk16(bp[0], bp[1]), acc[nt]);
        }
        __syncthreads();   // everyone done with bsh[cur] before it is re-staged
    }

#pragma unroll
    for (int nt = 0; nt < 16; ++nt) {
        const int col = nt * 16 + l16;
#pragma unroll
        for (int r = 0; r < 8; ++r) {
            const int n = mb * 128 + w * 16 + (hiN ? 8 : 0) + r;
            atomicAdd(&hacc[n * 256 + col], acc[nt][r]);
        }
    }
}

// ---------------------------------------------------------------------------
// k_finish: out[n] = relu(hacc[n,:]+bh) @ Wo + bo ; one wave per row.
// ---------------------------------------------------------------------------
__global__ void k_finish(const float* __restrict__ hacc, const float* __restrict__ bh,
                         const float* __restrict__ Wo, const float* __restrict__ bo,
                         float* __restrict__ out) {
    const int t = threadIdx.x, w = t >> 5, lane = t & 31;
    const int n = blockIdx.x * 8 + w;
    const float* hr = hacc + n * 256;
    float v = 0.f;
#pragma unroll
    for (int j = 0; j < 8; ++j) {
        const int c = lane + j * 32;
        const float h = hr[c] + bh[c];
        v += (h > 0.f ? h : 0.f) * Wo[c];
    }
#pragma unroll
    for (int m2 = 1; m2 < 32; m2 <<= 1) v += __shfl_xor(v, m2, 32);
    if (lane == 0) out[n] = v + bo[0];
}

// ---------------------------------------------------------------------------
extern "C" void kernel_launch(void* const* d_in, const int* in_sizes, int n_in,
                              void* d_out, int out_size, void* d_ws, size_t ws_size,
                              hipStream_t stream) {
    (void)in_sizes; (void)n_in; (void)out_size; (void)ws_size;

    const float* arr = (const float*)d_in[0];
    const float* Win = (const float*)d_in[1];
    const float* bin = (const float*)d_in[2];
    const float* Wf  = (const float*)d_in[3];
    const float* bf  = (const float*)d_in[4];
    const float* Vf  = (const float*)d_in[5];
    /* bv = d_in[6]: softmax is shift-invariant, mathematically a no-op */
    const float* lng = (const float*)d_in[7];
    const float* lnb = (const float*)d_in[8];
    const float* Wh  = (const float*)d_in[9];
    const float* bh  = (const float*)d_in[10];
    const float* Wo  = (const float*)d_in[11];
    const float* bo  = (const float*)d_in[12];
    float* out = (float*)d_out;

    // workspace layout
    const size_t XXB  = (size_t)NN * KFLAT * 2;        // 134,217,728
    const size_t WHFB = (size_t)KFLAT * HH * 2;        //  16,777,216
    const size_t WFFB = (size_t)DD * DD * 2;           //      32,768
    char* wsb = (char*)d_ws;
    _Float16* xx  = (_Float16*)(wsb);
    _Float16* whf = (_Float16*)(wsb + XXB);
    _Float16* wff = (_Float16*)(wsb + XXB + WHFB);
    float*    hacc = (float*)(wsb + XXB + WHFB + WFFB);  // 2,097,152 bytes

    hipMemsetAsync(hacc, 0, (size_t)NN * HH * sizeof(float), stream);

    // (32768*256 + 128*128) halves / 256 threads = 32832 blocks exactly
    k_convert<<<32832, 256, 0, stream>>>(Wh, Wf, whf, wff);

    k_attn<<<NN, 256, 0, stream>>>(arr, Win, bin, wff, bf, Vf, lng, lnb, xx);

    dim3 g2(16, 16);
    k_head<<<g2, 256, 0, stream>>>(xx, whf, hacc);

    k_finish<<<NN / 8, 256, 0, stream>>>(hacc, bh, Wo, bo, out);
}
